// MANN_14053132993101
// MI455X (gfx1250) — matrix operand order repaired and checked by reading, not compiled, not run on hardware
//
#include <hip/hip_runtime.h>
#include <math.h>
#include <stdint.h>

#if __has_include(<hip/amd_detail/amd_gfx1250_TDM.h>)
#define TDM_6ARG 1
#endif

#define B_TOTAL 131072
#define IN_DIM  256
#define HID     128
#define GH      64
#define NEXP    4
#define ADIM    3
#define EPS     1e-5f

#define ROWS_PER_WG 128
#define THREADS     256
#define NWAVES      8
#define NLOADS      13   // gw1, e1[0..3], e2[0..3], e3[0..3]

typedef __attribute__((ext_vector_type(16))) _Float16     v16h;
typedef __attribute__((ext_vector_type(8)))  float        v8f;
typedef __attribute__((ext_vector_type(4)))  unsigned int u32x4;
typedef __attribute__((ext_vector_type(4)))  int          i32x4;
typedef __attribute__((ext_vector_type(8)))  int          i32x8;
typedef _Float16     half_t;
typedef unsigned int u32;

// ---- f16 weight workspace layout (element offsets into d_ws) ----
#define WS_GW1   0
#define WS_E1    (GH * IN_DIM)                    // 16384
#define WS_E2    (WS_E1 + NEXP * HID * IN_DIM)    // 147456
#define WS_E3    (WS_E2 + NEXP * HID * HID)       // 212992 (e3 zero-padded 16x128)
#define WS_ELEMS (WS_E3 + NEXP * 16 * HID)        // 221184
#define WS_BYTES ((size_t)WS_ELEMS * 2)

// =====================================================================
// WMMA helpers
// =====================================================================
__device__ __forceinline__ v8f wmma16(v16h a, v16h b, v8f c) {
  return __builtin_amdgcn_wmma_f32_16x16x32_f16(
      false, a, false, b, (short)0, c, false, false);
}

// A 16x32 f16 fragment from row-major LDS tile
__device__ __forceinline__ v16h frag_A(const half_t* base, int row0, int k0,
                                       int ld, int lane) {
  int r  = row0 + (lane & 15);
  int kk = k0 + ((lane >> 4) << 3);
  const half_t* p = base + r * ld + kk;
  v16h a;
  ((uint4*)&a)[0] = *(const uint4*)(p);
  ((uint4*)&a)[1] = *(const uint4*)(p + 16);
  return a;
}

// B 32x16 f16 fragment; B[k][n] = W[n][k] with weights row-major [out][in]
__device__ __forceinline__ v16h frag_B(const half_t* wbase, int n0, int k0,
                                       int ld, int lane) {
  int n  = n0 + (lane & 15);
  int kk = k0 + ((lane >> 4) << 3);
  const half_t* p = wbase + n * ld + kk;
  v16h b;
  ((uint4*)&b)[0] = *(const uint4*)(p);
  ((uint4*)&b)[1] = *(const uint4*)(p + 16);
  return b;
}

__device__ __forceinline__ float hsum16(float x) {
  x += __shfl_xor(x, 1, 32);
  x += __shfl_xor(x, 2, 32);
  x += __shfl_xor(x, 4, 32);
  x += __shfl_xor(x, 8, 32);
  return x;
}

__device__ __forceinline__ void stage_f16(half_t* dst, const float* src, int n,
                                          int tid) {
  const float4* s4 = (const float4*)src;
  for (int i = tid; i < (n >> 2); i += THREADS) {
    float4 v = s4[i];
    half_t* d = dst + (i << 2);
    d[0] = (half_t)v.x; d[1] = (half_t)v.y;
    d[2] = (half_t)v.z; d[3] = (half_t)v.w;
  }
}

__device__ __forceinline__ void ln128_relu_store(v8f racc[8], const float* gv,
                                                 const float* bv, half_t* dst,
                                                 int wrow0, int hi, int c16) {
  float mu[8], inv[8];
#pragma unroll
  for (int v = 0; v < 8; ++v) {
    float s = 0.f, s2 = 0.f;
#pragma unroll
    for (int nt = 0; nt < 8; ++nt) { float f = racc[nt][v]; s += f; s2 += f * f; }
    s = hsum16(s); s2 = hsum16(s2);
    float m = s * (1.0f / 128.0f);
    float var = s2 * (1.0f / 128.0f) - m * m;
    mu[v] = m; inv[v] = rsqrtf(var + EPS);
  }
#pragma unroll
  for (int nt = 0; nt < 8; ++nt) {
    int col = nt * 16 + c16;
    float g = gv[col], bb = bv[col];
#pragma unroll
    for (int v = 0; v < 8; ++v) {
      float val = (racc[nt][v] - mu[v]) * inv[v] * g + bb;
      val = fmaxf(val, 0.0f);
      dst[(wrow0 + v + hi * 8) * HID + col] = (half_t)val;
    }
  }
}

// =====================================================================
// Tensor Data Mover: 1-D contiguous copy, global f16 -> LDS.
// D# group0: count=1, lds_addr, global_addr, type=2.
// D# group1: data_size=8B, tensor_dim0=tile_dim0=nbytes/8, dims1..=1.
// =====================================================================
__device__ __forceinline__ u32 lds_lo32(const void* p) {
  return (u32)(uintptr_t)p;   // flat LDS address: low 32 bits = LDS byte offset
}

__device__ __forceinline__ void tdm_load_1d(u32 lds_byte, const void* gsrc,
                                            u32 nbytes) {
  unsigned long long ga = (unsigned long long)(uintptr_t)gsrc;
  u32 n8 = nbytes >> 3;                         // 8-byte elements
  u32x4 g0;
  g0[0] = 1u;                                   // count=1, user descriptor
  g0[1] = lds_byte;                             // lds_addr [63:32]
  g0[2] = (u32)ga;                              // global_addr lo
  g0[3] = (u32)(ga >> 32) | 0x80000000u;        // global_addr hi | type=2
  i32x8 g1;
  g1[0] = (int)(3u << 16);                      // workgroup_mask=0, data_size=8B
  g1[1] = (int)((n8 & 0xffffu) << 16);          // tensor_dim0 lo16
  g1[2] = (int)(((n8 >> 16) & 0xffffu) | (1u << 16)); // dim0 hi16 | tensor_dim1=1
  g1[3] = (int)((n8 & 0xffffu) << 16);          // tile_dim0 = n8
  g1[4] = 1;                                    // tile_dim1=1, tile_dim2=0
  g1[5] = (int)n8;                              // tensor_dim0_stride lo32
  g1[6] = 0;                                    // stride0 hi | stride1 lo
  g1[7] = 0;                                    // stride1 hi
  i32x4 z4 = {0, 0, 0, 0};
#ifdef TDM_6ARG
  i32x8 z8 = {0, 0, 0, 0, 0, 0, 0, 0};
  __builtin_amdgcn_tensor_load_to_lds(g0, g1, z4, z4, z8, 0);
#else
  __builtin_amdgcn_tensor_load_to_lds(g0, g1, z4, z4, 0);
#endif
}

// staged-load table: index -> (src offset in ws, bytes)
__device__ __forceinline__ const half_t* load_src(const half_t* wsf, int li,
                                                  u32* bytes) {
  if (li == 0) { *bytes = GH * IN_DIM * 2; return wsf + WS_GW1; }
  if (li <= 4) { *bytes = HID * IN_DIM * 2; return wsf + WS_E1 + (li - 1) * HID * IN_DIM; }
  if (li <= 8) { *bytes = HID * HID * 2;   return wsf + WS_E2 + (li - 5) * HID * HID; }
  *bytes = 16 * HID * 2;                   return wsf + WS_E3 + (li - 9) * 16 * HID;
}

// fallback: stage load li directly from f32 globals (convert in VGPRs)
__device__ __forceinline__ void stage_load_sync(int li, half_t* dst,
                                                const float* gw1,
                                                const float* e1_w,
                                                const float* e2_w,
                                                const float* e3_w, int tid) {
  if (li == 0) {
    stage_f16(dst, gw1, GH * IN_DIM, tid);
  } else if (li <= 4) {
    stage_f16(dst, e1_w + (size_t)(li - 1) * HID * IN_DIM, HID * IN_DIM, tid);
  } else if (li <= 8) {
    stage_f16(dst, e2_w + (size_t)(li - 5) * HID * HID, HID * HID, tid);
  } else {
    int k = li - 9;
    for (int i = tid; i < 16 * HID; i += THREADS) {
      int r = i >> 7, c = i & 127;
      dst[i] = (r < ADIM) ? (half_t)e3_w[(size_t)(k * ADIM + r) * HID + c]
                          : (half_t)0.0f;
    }
  }
}

// =====================================================================
// prep kernel: convert all weights to f16 into d_ws (e3 zero-padded 16x128)
// =====================================================================
__global__ void prep_weights(const float* __restrict__ gw1,
                             const float* __restrict__ e1_w,
                             const float* __restrict__ e2_w,
                             const float* __restrict__ e3_w,
                             half_t* __restrict__ ws) {
  int idx = blockIdx.x * blockDim.x + threadIdx.x;
  int stride = gridDim.x * blockDim.x;
  for (int i = idx; i < GH * IN_DIM; i += stride)
    ws[WS_GW1 + i] = (half_t)gw1[i];
  for (int i = idx; i < NEXP * HID * IN_DIM; i += stride)
    ws[WS_E1 + i] = (half_t)e1_w[i];
  for (int i = idx; i < NEXP * HID * HID; i += stride)
    ws[WS_E2 + i] = (half_t)e2_w[i];
  for (int i = idx; i < NEXP * 16 * HID; i += stride) {
    int k = i >> 11, r = (i >> 7) & 15, c = i & 127;
    ws[WS_E3 + i] = (r < ADIM) ? (half_t)e3_w[(size_t)(k * ADIM + r) * HID + c]
                               : (half_t)0.0f;
  }
}

// =====================================================================
// fused main kernel (TDM double-buffered weights, or sync fallback)
// =====================================================================
template <bool TDM>
__global__ __launch_bounds__(THREADS, 1)
void mann_fused(const float* __restrict__ state,
                const float* __restrict__ gw1,  const float* __restrict__ gb1,
                const float* __restrict__ gln_g,const float* __restrict__ gln_b,
                const float* __restrict__ gw2,  const float* __restrict__ gb2,
                const float* __restrict__ temperature,
                const float* __restrict__ e1_w, const float* __restrict__ e1_b,
                const float* __restrict__ ln1_g,const float* __restrict__ ln1_b,
                const float* __restrict__ e2_w, const float* __restrict__ e2_b,
                const float* __restrict__ ln2_g,const float* __restrict__ ln2_b,
                const float* __restrict__ e3_w, const float* __restrict__ e3_b,
                const half_t* __restrict__ wsf,
                float* __restrict__ out_policy, float* __restrict__ out_blend) {
  // LDS layout (231424 B):
  //   [0)      sState : 128x256 f16 (65536) -> later reused as x2 (128x128)
  //   [65536)  sW0    : 65536   weight double-buffer 0
  //   [131072) sW1    : 65536   weight double-buffer 1
  //   [196608) sH/sX  : union gating h f32 | x1 f16 (32768)
  //   [229376) sBlend : 128x4 f32 (2048)
  extern __shared__ char smem[];
  half_t* sState = (half_t*)smem;
  half_t* sW0    = (half_t*)(smem + 65536);
  half_t* sW1    = (half_t*)(smem + 131072);
  float*  sH     = (float*) (smem + 196608);
  half_t* sX     = (half_t*)(smem + 196608);
  float*  sBlend = (float*) (smem + 229376);
  half_t* sWb[2] = {sW0, sW1};

  const int tid  = threadIdx.x;
  const int lane = tid & 31;
  const int wave = tid >> 5;
  const int hi   = lane >> 4;
  const int c16  = lane & 15;
  const int rowbase = blockIdx.x * ROWS_PER_WG;
  const int wrow0   = wave * 16;

  // step protocol: sync(prev compute done) -> issue next / stage cur ->
  //                wait own slice -> sync(all slices landed) -> compute
  auto step_begin = [&](int c) {
    __syncthreads();
    if (TDM) {
      if (c + 1 < NLOADS) {
        u32 bytes; const half_t* src = load_src(wsf, c + 1, &bytes);
        u32 slice = bytes >> 3;   // 8 waves, one slice each
        tdm_load_1d(lds_lo32(sWb[(c + 1) & 1]) + (u32)wave * slice,
                    (const char*)src + (size_t)wave * slice, slice);
        __builtin_amdgcn_s_wait_tensorcnt(1);
      } else {
        __builtin_amdgcn_s_wait_tensorcnt(0);
      }
    } else {
      stage_load_sync(c, sWb[c & 1], gw1, e1_w, e2_w, e3_w, tid);
    }
    __syncthreads();
  };

  // -------- preamble: kick load L0 (gw1), stage state tile ------------
  if (TDM) {
    u32 bytes; const half_t* src = load_src(wsf, 0, &bytes);
    u32 slice = bytes >> 3;
    tdm_load_1d(lds_lo32(sWb[0]) + (u32)wave * slice,
                (const char*)src + (size_t)wave * slice, slice);
  }
  stage_f16(sState, state + (size_t)rowbase * IN_DIM, ROWS_PER_WG * IN_DIM, tid);

  // -------- step 0: gating h = relu(LN(state @ gw1^T + gb1)) ----------
  step_begin(0);
  {
    v8f acc[4];
#pragma unroll
    for (int nt = 0; nt < 4; ++nt) { v8f z = {}; acc[nt] = z; }
    for (int ks = 0; ks < 8; ++ks) {
      v16h a = frag_A(sState, wrow0, ks * 32, IN_DIM, lane);
#pragma unroll
      for (int nt = 0; nt < 4; ++nt)
        acc[nt] = wmma16(a, frag_B(sWb[0], nt * 16, ks * 32, IN_DIM, lane), acc[nt]);
    }
#pragma unroll
    for (int nt = 0; nt < 4; ++nt) {
      float bv = gb1[nt * 16 + c16];
#pragma unroll
      for (int v = 0; v < 8; ++v) acc[nt][v] += bv;
    }
    float mu[8], inv[8];
#pragma unroll
    for (int v = 0; v < 8; ++v) {
      float s = 0.f, s2 = 0.f;
#pragma unroll
      for (int nt = 0; nt < 4; ++nt) { float f = acc[nt][v]; s += f; s2 += f * f; }
      s = hsum16(s); s2 = hsum16(s2);
      float m = s * (1.0f / 64.0f);
      float var = s2 * (1.0f / 64.0f) - m * m;
      mu[v] = m; inv[v] = rsqrtf(var + EPS);
    }
#pragma unroll
    for (int nt = 0; nt < 4; ++nt) {
      int col = nt * 16 + c16;
      float g = gln_g[col], bb = gln_b[col];
#pragma unroll
      for (int v = 0; v < 8; ++v) {
        float val = (acc[nt][v] - mu[v]) * inv[v] * g + bb;
        sH[(wrow0 + v + hi * 8) * GH + col] = fmaxf(val, 0.0f);
      }
    }
  }
  __syncthreads();

  // -------- gating part 2: logits -> softmax(blend) -------------------
  if (tid < ROWS_PER_WG) {
    const float* hr = sH + tid * GH;
    float lg[NEXP];
#pragma unroll
    for (int k = 0; k < NEXP; ++k) {
      float s = gb2[k];
      const float* wr = gw2 + k * GH;
      for (int j = 0; j < GH; ++j) s += hr[j] * wr[j];
      lg[k] = s;
    }
    float t = temperature[0];
    t = fminf(fmaxf(t, 0.1f), 2.0f);
    float m = fmaxf(fmaxf(lg[0], lg[1]), fmaxf(lg[2], lg[3]));
    float e[NEXP], Z = 0.f;
#pragma unroll
    for (int k = 0; k < NEXP; ++k) { e[k] = __expf((lg[k] - m) / t); Z += e[k]; }
    float rz = 1.0f / Z;
#pragma unroll
    for (int k = 0; k < NEXP; ++k) {
      float bv = e[k] * rz;
      sBlend[tid * NEXP + k] = bv;
      out_blend[(size_t)(rowbase + tid) * NEXP + k] = bv;
    }
  }

  // -------- steps 1..4: expert layer 1 (256 -> 128) -------------------
  v16h a1[8];
#pragma unroll
  for (int ks = 0; ks < 8; ++ks)
    a1[ks] = frag_A(sState, wrow0, ks * 32, IN_DIM, lane);

  v8f racc[8];
#pragma unroll
  for (int nt = 0; nt < 8; ++nt) { v8f z = {}; racc[nt] = z; }

  for (int k = 0; k < NEXP; ++k) {
    const int c = 1 + k;
    step_begin(c);
    const half_t* wbuf = sWb[c & 1];
    float bl[8];
#pragma unroll
    for (int v = 0; v < 8; ++v)
      bl[v] = sBlend[(wrow0 + v + hi * 8) * NEXP + k];
#pragma unroll
    for (int nt = 0; nt < 8; ++nt) {
      v8f acc = {};
#pragma unroll
      for (int ks = 0; ks < 8; ++ks)
        acc = wmma16(a1[ks], frag_B(wbuf, nt * 16, ks * 32, IN_DIM, lane), acc);
      float bvb = e1_b[k * HID + nt * 16 + c16];
#pragma unroll
      for (int v = 0; v < 8; ++v) racc[nt][v] += bl[v] * (acc[v] + bvb);
    }
  }
  ln128_relu_store(racc, ln1_g, ln1_b, sX, wrow0, hi, c16);

  // -------- steps 5..8: expert layer 2 (128 -> 128) -------------------
  v16h a2[4];
#pragma unroll
  for (int ks = 0; ks < 4; ++ks)
    a2[ks] = frag_A(sX, wrow0, ks * 32, HID, lane);
#pragma unroll
  for (int nt = 0; nt < 8; ++nt) { v8f z = {}; racc[nt] = z; }

  for (int k = 0; k < NEXP; ++k) {
    const int c = 5 + k;
    step_begin(c);
    const half_t* wbuf = sWb[c & 1];
    float bl[8];
#pragma unroll
    for (int v = 0; v < 8; ++v)
      bl[v] = sBlend[(wrow0 + v + hi * 8) * NEXP + k];
#pragma unroll
    for (int nt = 0; nt < 8; ++nt) {
      v8f acc = {};
#pragma unroll
      for (int ks = 0; ks < 4; ++ks)
        acc = wmma16(a2[ks], frag_B(wbuf, nt * 16, ks * 32, HID, lane), acc);
      float bvb = e2_b[k * HID + nt * 16 + c16];
#pragma unroll
      for (int v = 0; v < 8; ++v) racc[nt][v] += bl[v] * (acc[v] + bvb);
    }
  }
  half_t* sX2 = sState;   // reuse state tile as x2 (ld = HID)
  ln128_relu_store(racc, ln2_g, ln2_b, sX2, wrow0, hi, c16);

  // -------- steps 9..12: expert layer 3 (128 -> 3, N padded to 16) ----
  v16h a3[4];
#pragma unroll
  for (int ks = 0; ks < 4; ++ks)
    a3[ks] = frag_A(sX2, wrow0, ks * 32, HID, lane);
  v8f pol = {};

  for (int k = 0; k < NEXP; ++k) {
    const int c = 9 + k;
    step_begin(c);
    const half_t* wbuf = sWb[c & 1];
    float bl[8];
#pragma unroll
    for (int v = 0; v < 8; ++v)
      bl[v] = sBlend[(wrow0 + v + hi * 8) * NEXP + k];
    v8f acc = {};
#pragma unroll
    for (int ks = 0; ks < 4; ++ks)
      acc = wmma16(a3[ks], frag_B(wbuf, 0, ks * 32, HID, lane), acc);
    float bvb = (c16 < ADIM) ? e3_b[k * ADIM + c16] : 0.0f;
#pragma unroll
    for (int v = 0; v < 8; ++v) pol[v] += bl[v] * (acc[v] + bvb);
  }

  if (c16 < ADIM) {
#pragma unroll
    for (int v = 0; v < 8; ++v)
      out_policy[(size_t)(rowbase + wrow0 + v + hi * 8) * ADIM + c16] = pol[v];
  }
}

// =====================================================================
extern "C" void kernel_launch(void* const* d_in, const int* in_sizes, int n_in,
                              void* d_out, int out_size, void* d_ws,
                              size_t ws_size, hipStream_t stream) {
  (void)in_sizes; (void)n_in; (void)out_size;
  const float* state = (const float*)d_in[0];
  const float* gw1   = (const float*)d_in[1];
  const float* gb1   = (const float*)d_in[2];
  const float* gln_g = (const float*)d_in[3];
  const float* gln_b = (const float*)d_in[4];
  const float* gw2   = (const float*)d_in[5];
  const float* gb2   = (const float*)d_in[6];
  const float* temp  = (const float*)d_in[7];
  const float* e1_w  = (const float*)d_in[8];
  const float* e1_b  = (const float*)d_in[9];
  const float* ln1_g = (const float*)d_in[10];
  const float* ln1_b = (const float*)d_in[11];
  const float* e2_w  = (const float*)d_in[12];
  const float* e2_b  = (const float*)d_in[13];
  const float* ln2_g = (const float*)d_in[14];
  const float* ln2_b = (const float*)d_in[15];
  const float* e3_w  = (const float*)d_in[16];
  const float* e3_b  = (const float*)d_in[17];

  float* out_policy = (float*)d_out;
  float* out_blend  = (float*)d_out + (size_t)B_TOTAL * ADIM;
  half_t* wsf       = (half_t*)d_ws;

  const size_t smem_bytes = 231424;
  dim3 grid(B_TOTAL / ROWS_PER_WG);

  const bool use_tdm = (ws_size >= WS_BYTES);
  if (use_tdm) {
    prep_weights<<<256, THREADS, 0, stream>>>(gw1, e1_w, e2_w, e3_w, wsf);
    hipFuncSetAttribute((const void*)&mann_fused<true>,
                        hipFuncAttributeMaxDynamicSharedMemorySize,
                        (int)smem_bytes);
    mann_fused<true><<<grid, THREADS, smem_bytes, stream>>>(
        state, gw1, gb1, gln_g, gln_b, gw2, gb2, temp,
        e1_w, e1_b, ln1_g, ln1_b, e2_w, e2_b, ln2_g, ln2_b, e3_w, e3_b,
        wsf, out_policy, out_blend);
  } else {
    hipFuncSetAttribute((const void*)&mann_fused<false>,
                        hipFuncAttributeMaxDynamicSharedMemorySize,
                        (int)smem_bytes);
    mann_fused<false><<<grid, THREADS, smem_bytes, stream>>>(
        state, gw1, gb1, gln_g, gln_b, gw2, gb2, temp,
        e1_w, e1_b, ln1_g, ln1_b, e2_w, e2_b, ln2_g, ln2_b, e3_w, e3_b,
        wsf, out_policy, out_blend);
  }
}
